// T_LSTM_65489661329499
// MI455X (gfx1250) — matrix operand order repaired and checked by reading, not compiled, not run on hardware
//
#include <hip/hip_runtime.h>
#include <hip/hip_bf16.h>

typedef __attribute__((ext_vector_type(16))) _Float16 v16h;
typedef __attribute__((ext_vector_type(8)))  _Float16 v8h;
typedef __attribute__((ext_vector_type(8)))  float    v8f;

// ---------------------------------------------------------------------------
// helpers
// ---------------------------------------------------------------------------
__device__ __forceinline__ v8f wmma16(v16h a, v16h b, v8f c) {
    // D = A(16x32 f16) * B(32x16 f16) + C(16x16 f32)
    return __builtin_amdgcn_wmma_f32_16x16x32_f16(
        /*neg_a=*/false, a, /*neg_b=*/false, b,
        /*c_mod=*/(short)0, c, /*reuse_a=*/false, /*reuse_b=*/false);
}

// A-matrix 16x32 f16 fragment (contiguous K, matches B frags):
//   lanes 0-15 : VGPR0-7 = K 0..15   (aoff = 0)
//   lanes16-31 : VGPR0-7 = K 16..31  (aoff = 8)
// p points at row[kbase]; aoff = (lane>=16) ? 8 : 0.
__device__ __forceinline__ v16h load_a_frag(const _Float16* p, int aoff) {
    v8h lo = *(const v8h*)(p + 2 * aoff);
    v8h hi = *(const v8h*)(p + 2 * aoff + 8);
    v16h r;
#pragma unroll
    for (int i = 0; i < 8; ++i) { r[i] = lo[i]; r[i + 8] = hi[i]; }
    return r;
}

__device__ __forceinline__ float sig_(float x)  { return 1.0f / (1.0f + __expf(-x)); }
__device__ __forceinline__ float tanh_(float x) { return 1.0f - 2.0f / (__expf(2.0f * x) + 1.0f); }

// d_ws layout (halves):  WdT[128][128] @0 | WxT[512][64] @16384 | WhT[512][128] @49152
#define WXT_OFF 16384
#define WHT_OFF 49152
#define HSCR_BYTE_OFF 229376   // 114688 halves, 256B aligned

// ---------------------------------------------------------------------------
// 1) weight convert + transpose to f16, N-major (so B frags are contiguous)
// ---------------------------------------------------------------------------
__global__ void tlstm_prep_kernel(const float* __restrict__ Wd,
                                  const float* __restrict__ Wx,
                                  const float* __restrict__ Wh,
                                  _Float16* __restrict__ o) {
    int i = blockIdx.x * blockDim.x + threadIdx.x;
    if (i < 16384) {                       // WdT[n][k] = Wd[k][n], 128x128
        int n = i >> 7, k = i & 127;
        o[n * 128 + k] = (_Float16)Wd[k * 128 + n];
    } else if (i < 49152) {                // WxT[n][k] = Wx[k][n], n<512,k<64
        int j = i - 16384; int n = j >> 6, k = j & 63;
        o[WXT_OFF + n * 64 + k] = (_Float16)Wx[k * 512 + n];
    } else if (i < 114688) {               // WhT[n][k] = Wh[k][n], n<512,k<128
        int j = i - 49152; int n = j >> 7, k = j & 127;
        o[WHT_OFF + n * 128 + k] = (_Float16)Wh[k * 512 + n];
    }
}

// ---------------------------------------------------------------------------
// 2) T-LSTM scan: 16 batch rows per workgroup, wave w owns hidden cols 16w..16w+15
// ---------------------------------------------------------------------------
__global__ __launch_bounds__(256) void tlstm_scan_kernel(
    const float* __restrict__ x, const float* __restrict__ h0,
    const float* __restrict__ bd, const float* __restrict__ bg,
    const _Float16* __restrict__ wbuf, float* __restrict__ hout) {

    const _Float16* WdT = wbuf;
    const _Float16* WxT = wbuf + WXT_OFF;
    const _Float16* WhT = wbuf + WHT_OFF;

    __shared__ __align__(32) _Float16 sH[16 * 128];   // h (f16, A frags)
    __shared__ __align__(32) _Float16 sC[16 * 128];   // c (f16, A frags)
    __shared__ __align__(32) float    sCf[16 * 128];  // c (f32 state)
    __shared__ __align__(32) float    sHf[16 * 128];  // h (f32 state)
    __shared__ __align__(32) _Float16 sF[16 * 64];    // feat_t (f16)
    __shared__ float sG[16];                          // g = 1/log(e+dt) per row

    const int tid   = threadIdx.x;
    const int lane  = tid & 31;
    const int wv    = tid >> 5;
    const int rbase = blockIdx.x * 16;

    // init h = c = broadcast(h0)
    for (int i = tid; i < 16 * 128; i += 256) {
        float v = h0[i & 127];
        sHf[i] = v; sCf[i] = v;
        sH[i] = (_Float16)v; sC[i] = (_Float16)v;
    }
    __syncthreads();

    const int arow    = lane & 15;            // A-frag row
    const int aoff    = (lane >> 4) << 3;     // A-frag K sub-offset (0/8)
    const int boff    = (lane >> 4) << 4;     // B-frag K sub-offset (0/16)
    const int colA    = wv * 16 + (lane & 15);// this lane's hidden column
    const int rowbase = (lane >> 4) << 3;     // C-tile row offset (0/8)

    const float biasd = bd[colA];
    const float bi = bg[colA], bf = bg[128 + colA], bo = bg[256 + colA], bc = bg[384 + colA];

    for (int t = 0; t < 64; ++t) {
        // ---- stage x_t: feat -> f16, dt -> g -------------------------------
        for (int i = tid; i < 16 * 64; i += 256) {
            int r = i >> 6, k = i & 63;
            sF[i] = (_Float16)x[((size_t)(rbase + r) * 64 + t) * 65 + 1 + k];
        }
        if (tid < 16) {
            float dt = x[((size_t)(rbase + tid) * 64 + t) * 65];
            sG[tid] = 1.0f / __logf(2.718281828459045f + dt);
        }
        __syncthreads();

        // ---- cs = tanh(c @ Wd + bd), 16x16 tile ----------------------------
        v8f accd = {};
#pragma unroll
        for (int ks = 0; ks < 4; ++ks) {
            v16h a  = load_a_frag(&sC[arow * 128 + ks * 32], aoff);
            v16h bm = *(const v16h*)(WdT + (size_t)colA * 128 + ks * 32 + boff);
            accd = wmma16(a, bm, accd);
        }

        // ---- gates = feat@Wx + h@Wh; i/f/o/ct tiles co-located per wave ----
        v8f gi = {}, gf = {}, go = {}, gc = {};
#pragma unroll
        for (int ks = 0; ks < 2; ++ks) {      // K over FEAT=64
            v16h a = load_a_frag(&sF[arow * 64 + ks * 32], aoff);
            const _Float16* bp = WxT + (size_t)colA * 64 + ks * 32 + boff;
            gi = wmma16(a, *(const v16h*)(bp        ), gi);
            gf = wmma16(a, *(const v16h*)(bp +  8192), gf);
            go = wmma16(a, *(const v16h*)(bp + 16384), go);
            gc = wmma16(a, *(const v16h*)(bp + 24576), gc);
        }
#pragma unroll
        for (int ks = 0; ks < 4; ++ks) {      // K over HIDDEN=128
            v16h a = load_a_frag(&sH[arow * 128 + ks * 32], aoff);
            const _Float16* bp = WhT + (size_t)colA * 128 + ks * 32 + boff;
            gi = wmma16(a, *(const v16h*)(bp        ), gi);
            gf = wmma16(a, *(const v16h*)(bp + 16384), gf);
            go = wmma16(a, *(const v16h*)(bp + 32768), go);
            gc = wmma16(a, *(const v16h*)(bp + 49152), gc);
        }

        // ---- nonlinear state update (C-tile layout: row=v+8*(lane>=16)) ----
        float cn[8], hn[8];
#pragma unroll
        for (int e = 0; e < 8; ++e) {
            int   row   = e + rowbase;
            float cprev = sCf[row * 128 + colA];
            float g     = sG[row];
            float cs    = tanh_(accd[e] + biasd);
            float cadj  = cprev - cs + cs * g;
            float vi    = sig_(gi[e] + bi);
            float vf    = sig_(gf[e] + bf);
            float vo    = sig_(go[e] + bo);
            float vc    = tanh_(gc[e] + bc);
            float c2    = vf * cadj + vi * vc;
            cn[e] = c2;
            hn[e] = vo * tanh_(c2);
        }
        __syncthreads();                       // everyone done reading sH/sC
#pragma unroll
        for (int e = 0; e < 8; ++e) {
            int row = e + rowbase;
            sCf[row * 128 + colA] = cn[e];
            sC [row * 128 + colA] = (_Float16)cn[e];
            sHf[row * 128 + colA] = hn[e];
            sH [row * 128 + colA] = (_Float16)hn[e];
        }
        __syncthreads();
    }

    // final hidden state -> scratch
    for (int i = tid; i < 16 * 128; i += 256)
        hout[(size_t)rbase * 128 + i] = sHf[i];
}

// ---------------------------------------------------------------------------
// 3) MLP head: relu(h@W1+b1) -> relu(@W2+b2) -> softmax(@W3+b3)
//    one thread per batch row; weight indices are wave-uniform -> s_loads
// ---------------------------------------------------------------------------
__global__ __launch_bounds__(128) void tlstm_mlp_kernel(
    const float* __restrict__ h,
    const float* __restrict__ W1, const float* __restrict__ b1,
    const float* __restrict__ W2, const float* __restrict__ b2,
    const float* __restrict__ W3, const float* __restrict__ b3,
    float* __restrict__ out) {

    int bidx = blockIdx.x * blockDim.x + threadIdx.x;
    const float* hv = h + (size_t)bidx * 128;
    float hl[128];
#pragma unroll
    for (int k = 0; k < 128; ++k) hl[k] = hv[k];

    float z2[32];
#pragma unroll
    for (int j = 0; j < 32; ++j) z2[j] = b2[j];

    for (int n = 0; n < 256; ++n) {
        float s = b1[n];
#pragma unroll
        for (int k = 0; k < 128; ++k) s += hl[k] * W1[k * 256 + n];
        s = fmaxf(s, 0.0f);
#pragma unroll
        for (int j = 0; j < 32; ++j) z2[j] += s * W2[n * 32 + j];
    }

    float z3[2];
#pragma unroll
    for (int j2 = 0; j2 < 2; ++j2) {
        float s = b3[j2];
#pragma unroll
        for (int j = 0; j < 32; ++j) s += fmaxf(z2[j], 0.0f) * W3[j * 2 + j2];
        z3[j2] = s;
    }
    float m   = fmaxf(z3[0], z3[1]);
    float e0  = __expf(z3[0] - m), e1 = __expf(z3[1] - m);
    float inv = 1.0f / (e0 + e1);
    out[bidx * 2 + 0] = e0 * inv;
    out[bidx * 2 + 1] = e1 * inv;
}

// ---------------------------------------------------------------------------
extern "C" void kernel_launch(void* const* d_in, const int* in_sizes, int n_in,
                              void* d_out, int out_size, void* d_ws, size_t ws_size,
                              hipStream_t stream) {
    const float* x  = (const float*)d_in[0];
    const float* h0 = (const float*)d_in[1];
    const float* Wd = (const float*)d_in[2];
    const float* bd = (const float*)d_in[3];
    const float* Wx = (const float*)d_in[4];
    const float* Wh = (const float*)d_in[5];
    const float* bg = (const float*)d_in[6];
    const float* W1 = (const float*)d_in[7];
    const float* b1 = (const float*)d_in[8];
    const float* W2 = (const float*)d_in[9];
    const float* b2 = (const float*)d_in[10];
    const float* W3 = (const float*)d_in[11];
    const float* b3 = (const float*)d_in[12];

    _Float16* wbuf = (_Float16*)d_ws;
    float*    hscr = (float*)((char*)d_ws + HSCR_BYTE_OFF);  // 8192*128 f32

    tlstm_prep_kernel<<<448, 256, 0, stream>>>(Wd, Wx, Wh, wbuf);
    tlstm_scan_kernel<<<512, 256, 0, stream>>>(x, h0, bd, bg, wbuf, hscr);
    tlstm_mlp_kernel<<<64, 128, 0, stream>>>(hscr, W1, b1, W2, b2, W3, b3,
                                             (float*)d_out);
}
